// LogSinkhorn_43791486550190
// MI455X (gfx1250) — matrix operand order repaired and checked by reading, not compiled, not run on hardware
//
#include <hip/hip_runtime.h>
#include <cstdint>
#include <cstddef>

// ---------------------------------------------------------------------------
// Log-domain Sinkhorn, B=16, N=M=2048, eps=0.05, 50 iterations.
//
// Math: the reference's logsumexp updates are exactly u = r/(K v),
// v = c/(K^T u) with K = exp(-cost/eps) in [2e-9, 1] -- safe in f32/f16,
// so we run plain Sinkhorn on a materialized fp16 K.
//
// Data movement: fp16 K is 128 MB -> resident in the 192 MB L2, so the
// 100 mat-vec passes stream from L2 instead of pulling 12.8 GB from HBM.
//
// Row pass (u-update): v_wmma_f32_16x16x32_f16, one wave per (16-row block,
// quarter of k), 4-way k-split combined in LDS -> 8192 waves of latency
// hiding. Column pass: coalesced deterministic LDS reduction, 1024 blocks.
// ---------------------------------------------------------------------------

typedef __attribute__((ext_vector_type(16))) _Float16 v16h;
typedef __attribute__((ext_vector_type(8)))  float    v8f;
typedef __attribute__((ext_vector_type(4)))  _Float16 v4h;
typedef __attribute__((ext_vector_type(4)))  float    v4f;

constexpr int   SB = 16;       // batches
constexpr int   SN = 2048;     // source dim
constexpr int   SM = 2048;     // target dim
constexpr float INV_EPS = 20.0f;   // 1/0.05
constexpr int   ITERS = 50;

// ---------------------------------------------------------------- pack ----
// Kh[b,i,j] = fp16(exp(-cost/eps)), vectorized 4-wide.
__global__ void sink_pack(const float* __restrict__ cost,
                          _Float16* __restrict__ Kh) {
  const long total4 = (long)SB * SN * SM / 4;
  long idx = (long)blockIdx.x * blockDim.x + threadIdx.x;
  const long stride = (long)gridDim.x * blockDim.x;
  const v4f* c4 = (const v4f*)cost;
  v4h* k4 = (v4h*)Kh;
  for (; idx < total4; idx += stride) {
    v4f c = c4[idx];
    v4h h;
    h[0] = (_Float16)__expf(-c[0] * INV_EPS);
    h[1] = (_Float16)__expf(-c[1] * INV_EPS);
    h[2] = (_Float16)__expf(-c[2] * INV_EPS);
    h[3] = (_Float16)__expf(-c[3] * INV_EPS);
    k4[idx] = h;
  }
}

// ---------------------------------------------------------------- init ----
// log_v0 = 0  ->  v0 = 1.
__global__ void sink_init_v(float* __restrict__ v, _Float16* __restrict__ vh) {
  int i = blockIdx.x * blockDim.x + threadIdx.x;
  if (i < SB * SM) { v[i] = 1.0f; vh[i] = (_Float16)1.0f; }
}

// ------------------------------------------------------------- u-update ----
// u[b,row] = (r+1e-12) / sum_j K[b,row,j]*v[b,j].
// Block = 256 threads = 8 waves = 2 row-blocks x 4 k-splits.
// Each wave: 16 of the 64 k-chunks (512 k values), WMMA-accumulated.
// A fragment (16x32 f16) per ISA layout:
//   lane l (m = l&15, hi = l>=16), VGPR p: k0 = 2p + (p>=4 ? 8:0) + hi*8
//   -> dword index within k-chunk: p + (p>=4?4:0) + hi*4   (one 64B line/row)
// B fragment (32x16 f16): VGPR q: k = 2q + (q>=4 ? 8:0) + hi*8 -> dword q + (q>=4?4:0) + hi*4;
//   v chunk replicated across all 16 columns (broadcast loads), so every
//   D column holds the same dot-product partial.
__global__ void __launch_bounds__(256)
sink_update_u(const _Float16* __restrict__ Kh,
              const _Float16* __restrict__ vh,
              const float* __restrict__ r,
              float* __restrict__ u) {
  const int lane = threadIdx.x & 31;
  const int hi   = lane >> 4;                 // 0 or 1
  const int wave = threadIdx.x >> 5;
  const int wgrb = wave >> 2;                 // row-block within block (0,1)
  const int ks   = wave & 3;                  // k-split (0..3)
  const int w    = blockIdx.x * 2 + wgrb;     // global row-block: [0, B*N/16)
  const int b    = w >> 7;                    // 128 row-blocks per batch
  const int rb   = w & 127;
  const int row  = rb * 16 + (lane & 15);

  const uint32_t* Krow =
      (const uint32_t*)(Kh + ((size_t)b * SN + row) * (size_t)SM);
  const uint32_t* Vrow = (const uint32_t*)(vh + (size_t)b * SM);

  const int hi4 = hi ? 4 : 0;
  const int hi8 = hi ? 8 : 0;

  v8f acc = {};
  const int kc0 = ks * 16;                    // 16 chunks of 32 k each
  for (int kc = kc0; kc < kc0 + 16; ++kc) {
    const int base = kc * 16;  // dword base of this 32-wide k chunk
    union { uint32_t u32[8]; v16h h; } A, Bf;
#pragma unroll
    for (int p = 0; p < 8; ++p)
      A.u32[p] = Krow[base + p + ((p >= 4) ? 4 : 0) + hi4];
#pragma unroll
    for (int q = 0; q < 8; ++q)
      Bf.u32[q] = Vrow[base + q + ((q >= 4) ? 4 : 0) + hi4];
    acc = __builtin_amdgcn_wmma_f32_16x16x32_f16(
        /*neg_a=*/false, A.h, /*neg_b=*/false, Bf.h,
        /*c_mod=*/(short)0, acc, /*reuse_a=*/false, /*reuse_b=*/false);
  }

  // D layout: VGPR p holds M=p (lanes 0-15) / M=p+8 (lanes 16-31); all
  // columns identical, so lanes 0 and 16 deposit the 16 row partials.
  __shared__ float part[2][4][16];            // [row-block][k-split][row]
  if ((lane & 15) == 0) {
#pragma unroll
    for (int p = 0; p < 8; ++p) part[wgrb][ks][p + hi8] = acc[p];
  }
  __syncthreads();

  if (threadIdx.x < 32) {
    const int rbsel = threadIdx.x >> 4;       // which row-block
    const int rr    = threadIdx.x & 15;       // row within block
    const float s = part[rbsel][0][rr] + part[rbsel][1][rr] +
                    part[rbsel][2][rr] + part[rbsel][3][rr];
    const int w2  = blockIdx.x * 2 + rbsel;
    const int b2  = w2 >> 7;
    const int rb2 = w2 & 127;
    const size_t idx = (size_t)b2 * SN + rb2 * 16 + rr;
    u[idx] = (r[idx] + 1e-12f) / s;
  }
}

// ------------------------------------------------------------- v-update ----
// v[b,j] = (c+1e-12) / sum_i K[b,i,j]*u[b,i].
// Block = (batch, 32-column slab). Thread t: column pair (t&15)*2, row
// 1/16th t>>4 (128 rows each). Per row-step the 16 active pairs read 64B
// contiguous per row group. Deterministic LDS tree combine (no atomics).
__global__ void __launch_bounds__(256)
sink_update_v(const _Float16* __restrict__ Kh,
              const float* __restrict__ u,
              const float* __restrict__ c,
              float* __restrict__ v,
              _Float16* __restrict__ vh) {
  const int b   = blockIdx.x >> 6;   // 64 column slabs per batch
  const int jb  = blockIdx.x & 63;
  const int t   = threadIdx.x;
  const int cp  = t & 15;            // column pair within slab
  const int seg = t >> 4;            // row segment 0..15
  const int j0  = jb * 32 + cp * 2;

  const uint32_t* Kd = (const uint32_t*)(Kh + (size_t)b * SN * SM);
  const float* ub = u + (size_t)b * SN;

  float a0 = 0.0f, a1 = 0.0f;
  const int rbeg = seg * (SN / 16), rend = rbeg + SN / 16;
  for (int i = rbeg; i < rend; ++i) {
    union { uint32_t w; _Float16 h[2]; } cv;
    cv.w = Kd[(size_t)i * (SM / 2) + (j0 >> 1)];
    const float ui = ub[i];
    a0 = fmaf((float)cv.h[0], ui, a0);
    a1 = fmaf((float)cv.h[1], ui, a1);
  }

  __shared__ float red0[16][16];
  __shared__ float red1[16][16];
  red0[seg][cp] = a0;
  red1[seg][cp] = a1;
  __syncthreads();

  if (t < 16) {
    float s0 = 0.0f, s1 = 0.0f;
#pragma unroll
    for (int o = 0; o < 16; ++o) { s0 += red0[o][t]; s1 += red1[o][t]; }
    const int j = jb * 32 + t * 2;
    const size_t idx = (size_t)b * SM + j;
    const float v0 = (c[idx]     + 1e-12f) / s0;
    const float v1 = (c[idx + 1] + 1e-12f) / s1;
    v[idx] = v0;            v[idx + 1] = v1;
    vh[idx] = (_Float16)v0; vh[idx + 1] = (_Float16)v1;
  }
}

// ----------------------------------------------------------------- plan ----
// T[b,i,j] = u[b,i] * K[b,i,j] * v[b,j], vectorized 4-wide along j.
__global__ void sink_plan(const _Float16* __restrict__ Kh,
                          const float* __restrict__ u,
                          const float* __restrict__ v,
                          float* __restrict__ T) {
  const long total4 = (long)SB * SN * SM / 4;
  long idx = (long)blockIdx.x * blockDim.x + threadIdx.x;
  const long stride = (long)gridDim.x * blockDim.x;
  const v4h* K4 = (const v4h*)Kh;
  v4f* T4 = (v4f*)T;
  for (; idx < total4; idx += stride) {
    const long e0 = idx * 4;
    const int b   = (int)(e0 / ((long)SN * SM));
    const long rm = e0 - (long)b * SN * SM;
    const int i   = (int)(rm / SM);
    const int j   = (int)(rm % SM);
    const v4h k   = K4[idx];
    const float uu = u[(size_t)b * SN + i];
    const float* vb = v + (size_t)b * SM + j;
    v4f o;
    o[0] = uu * (float)k[0] * vb[0];
    o[1] = uu * (float)k[1] * vb[1];
    o[2] = uu * (float)k[2] * vb[2];
    o[3] = uu * (float)k[3] * vb[3];
    T4[idx] = o;
  }
}

// --------------------------------------------------------------- launch ----
extern "C" void kernel_launch(void* const* d_in, const int* in_sizes, int n_in,
                              void* d_out, int out_size, void* d_ws,
                              size_t ws_size, hipStream_t stream) {
  const float* cost = (const float*)d_in[0];
  const float* rmar = (const float*)d_in[1];
  const float* cmar = (const float*)d_in[2];

  // Workspace layout (~128.3 MB):
  //   [0, 128MB)        fp16 K           (L2-resident working set)
  //   + B*N*4           u (f32)
  //   + B*M*4           v (f32)
  //   + B*M*2           v (f16, packed for WMMA B-fragment loads)
  char* ws = (char*)d_ws;
  _Float16* Kh = (_Float16*)ws;
  float*    u  = (float*)(ws + (size_t)SB * SN * SM * 2);
  float*    v  = (float*)((char*)u + (size_t)SB * SN * 4);
  _Float16* vh = (_Float16*)((char*)v + (size_t)SB * SM * 4);

  sink_pack<<<4096, 256, 0, stream>>>(cost, Kh);
  sink_init_v<<<(SB * SM + 255) / 256, 256, 0, stream>>>(v, vh);

  for (int it = 0; it < ITERS; ++it) {
    // u = r / (K v)   (WMMA row pass; 8192 waves, 4-way k-split)
    sink_update_u<<<(SB * (SN / 16)) / 2, 256, 0, stream>>>(Kh, vh, rmar, u);
    // v = c / (K^T u) (coalesced column pass; 1024 blocks)
    sink_update_v<<<SB * 64, 256, 0, stream>>>(Kh, u, cmar, v, vh);
  }

  sink_plan<<<8192, 256, 0, stream>>>(Kh, u, v, (float*)d_out);
}
